// MultiHeadAttention_76209899700363
// MI455X (gfx1250) — matrix operand order repaired and checked by reading, not compiled, not run on hardware
//
#include <hip/hip_runtime.h>
#include <hip/hip_bf16.h>
#include <type_traits>

typedef __attribute__((ext_vector_type(16))) __bf16 v16bf;
typedef __attribute__((ext_vector_type(8)))  __bf16 v8bf;
typedef __attribute__((ext_vector_type(2)))  __bf16 v2bf;
typedef __attribute__((ext_vector_type(8)))  float  v8f;

constexpr int B_ = 4, T_ = 2048, C_ = 1024, H_ = 16, HD_ = 64;
// C^-0.5 * log2(e): fold attention scale AND base-2 conversion into Q once.
constexpr float SCALE_L2E = 0.03125f * 1.44269504088896340736f;

constexpr int KSTR = 72;   // Klds row stride (halves): 144B, 16B-aligned, bank-spread
constexpr int VSTR = 40;   // Vt col stride (halves): 80B, 16B-aligned, bank-spread
constexpr int PSTR = 40;   // Plds row stride (halves)

__device__ __forceinline__ float rowmax16(float v) {
#pragma unroll
  for (int m = 8; m >= 1; m >>= 1) v = fmaxf(v, __shfl_xor(v, m, 32));
  return v;
}
__device__ __forceinline__ float rowsum16(float v) {
#pragma unroll
  for (int m = 8; m >= 1; m >>= 1) v += __shfl_xor(v, m, 32);
  return v;
}
// 16 contiguous bf16 from LDS as two aligned b128 loads.
__device__ __forceinline__ v16bf ld16(const __bf16* p) {
  v8bf lo = *(const v8bf*)p;
  v8bf hi = *(const v8bf*)(p + 8);
  return __builtin_shufflevector(lo, hi, 0, 1, 2, 3, 4, 5, 6, 7,
                                 8, 9, 10, 11, 12, 13, 14, 15);
}

// ---------------------------------------------------------------------------
// Flash attention: one wave = 16 query rows, block = 8 waves = 128 query rows.
// Key tiles of 32 in LDS (bf16): K row-major (padded), V transposed+packed so
// every WMMA fragment is contiguous 16B ds_load_b128s.
// ---------------------------------------------------------------------------
__global__ __launch_bounds__(256)
void flash_attn_kernel(const float* __restrict__ qg,
                       const float* __restrict__ kg,
                       const float* __restrict__ vg,
                       float* __restrict__ attn_out)
{
  __shared__ __bf16 Klds[32 * KSTR];       // [key][hd]
  __shared__ __bf16 Vt[64 * VSTR];         // [hd col][key pairs packed]
  __shared__ __bf16 Plds[8 * 16 * PSTR];   // per-wave P staging

  const int lane = threadIdx.x & 31;
  const int wv   = threadIdx.x >> 5;
  const int half = lane >> 4;
  const int ln   = lane & 15;
  __bf16* Pw = &Plds[wv * 16 * PSTR];

  const int b = blockIdx.z, h = blockIdx.y;
  const int hoff = h * HD_;
  const size_t bbase = (size_t)b * T_ * C_;
  const float* qb  = qg + bbase;
  const float* kgb = kg + bbase;
  const float* vgb = vg + bbase;

  const int q0         = blockIdx.x * 128 + wv * 16;
  const int q_hi_wave  = q0 + 15;
  const int q_hi_block = blockIdx.x * 128 + 127;

  // Q fragments (A-matrix, 16x32 bf16 per hd-chunk), scale pre-folded.
  v16bf aQ[2];
#pragma unroll
  for (int c = 0; c < 2; ++c) {
#pragma unroll
    for (int v = 0; v < 8; ++v) {
      const int kk = 2 * v + 16 * half;
      const float* p = qb + (size_t)(q0 + ln) * C_ + hoff + c * 32 + kk;
      aQ[c][2 * v]     = (__bf16)(p[0] * SCALE_L2E);
      aQ[c][2 * v + 1] = (__bf16)(p[1] * SCALE_L2E);
    }
  }

  v8f o[4] = {};
  float m_i[8], l_i[8], alpha[8];
#pragma unroll
  for (int r = 0; r < 8; ++r) { m_i[r] = -__builtin_inff(); l_i[r] = 0.f; }

  // cooperative-loader index precompute
  const int krow = threadIdx.x >> 3;          // 0..31
  const int kc0  = (threadIdx.x & 7) * 8;     // 0..56
  const int vrp  = threadIdx.x >> 4;          // 0..15 (key pair)
  const int vc0  = (threadIdx.x & 15) * 4;    // 0..60 (hd cols)

#pragma unroll 1
  for (int kbase = 0; kbase <= q_hi_block; kbase += 32) {
    // ---- cooperative K tile (row major, f32->bf16, b128 stores)
    {
      const float* kr = kgb + (size_t)(kbase + krow) * C_ + hoff + kc0;
#pragma unroll
      for (int j = 0; j < 8; ++j) Klds[krow * KSTR + kc0 + j] = (__bf16)kr[j];
      // ---- cooperative V tile, transposed + key-pair packed
      const float* vr0 = vgb + (size_t)(kbase + 2 * vrp) * C_ + hoff + vc0;
      const float* vr1 = vr0 + C_;
#pragma unroll
      for (int j = 0; j < 4; ++j) {
        v2bf pr;
        pr[0] = (__bf16)vr0[j];
        pr[1] = (__bf16)vr1[j];
        *(v2bf*)&Vt[(vc0 + j) * VSTR + 2 * vrp] = pr;
      }
      if (kbase + 32 <= q_hi_block) {   // prefetch next tile into caches
        __builtin_prefetch(kr + 32 * C_, 0, 3);
        __builtin_prefetch(vr0 + 32 * C_, 0, 3);
      }
    }
    __syncthreads();

    if (kbase <= q_hi_wave) {
      // ---- S = Q * K^T : 16 q-rows x 32 keys, accumulate over hd (2 x K=32)
      v8f s0 = {}, s1 = {};
#pragma unroll
      for (int c = 0; c < 2; ++c) {
        v16bf bK0 = ld16(&Klds[ln * KSTR + c * 32 + 16 * half]);
        v16bf bK1 = ld16(&Klds[(ln + 16) * KSTR + c * 32 + 16 * half]);
        s0 = __builtin_amdgcn_wmma_f32_16x16x32_bf16(false, aQ[c], false, bK0,
                                                     (short)0, s0, false, false);
        s1 = __builtin_amdgcn_wmma_f32_16x16x32_bf16(false, aQ[c], false, bK1,
                                                     (short)0, s1, false, false);
      }

      // ---- online softmax (base 2), mask only on diagonal-straddling tiles
      auto softmax_body = [&](auto mask_tag) {
        constexpr bool MASK = decltype(mask_tag)::value;
#pragma unroll
        for (int r = 0; r < 8; ++r) {
          float f0 = s0[r];
          float f1 = s1[r];
          if (MASK) {
            const int qrow = q0 + r + 8 * half;
            if (kbase + ln      > qrow) f0 = -__builtin_inff();
            if (kbase + 16 + ln > qrow) f1 = -__builtin_inff();
          }
          const float mx   = rowmax16(fmaxf(f0, f1));
          const float mnew = fmaxf(m_i[r], mx);
          const float a    = exp2f(m_i[r] - mnew);
          const float p0   = exp2f(f0 - mnew);
          const float p1   = exp2f(f1 - mnew);
          l_i[r]  = a * l_i[r] + rowsum16(p0 + p1);
          m_i[r]  = mnew;
          alpha[r] = a;
          const int m = r + 8 * half;
          Pw[m * PSTR + ln]      = (__bf16)p0;
          Pw[m * PSTR + ln + 16] = (__bf16)p1;
        }
      };
      if (kbase + 31 > q0) softmax_body(std::true_type{});
      else                 softmax_body(std::false_type{});

#pragma unroll
      for (int j = 0; j < 4; ++j)
#pragma unroll
        for (int r = 0; r < 8; ++r) o[j][r] *= alpha[r];

      // DS ops are in-order per wave; stop compiler reordering the re-layout.
      asm volatile("" ::: "memory");
      __builtin_amdgcn_wave_barrier();

      // ---- re-layout P into A-fragment (two aligned b128 chunks)
      v8bf plo = *(const v8bf*)&Pw[ln * PSTR + 16 * half];
      v8bf phi = *(const v8bf*)&Pw[ln * PSTR + 16 * half + 8];
      v16bf aP = __builtin_shufflevector(plo, phi, 0, 1, 2, 3, 4, 5, 6, 7,
                                         8, 9, 10, 11, 12, 13, 14, 15);

      // ---- O += P * V  (4 hd-chunks of 16 columns)
#pragma unroll
      for (int j = 0; j < 4; ++j) {
        v16bf bV = ld16(&Vt[(j * 16 + ln) * VSTR + 16 * half]);
        o[j] = __builtin_amdgcn_wmma_f32_16x16x32_bf16(false, aP, false, bV,
                                                       (short)0, o[j], false, false);
      }
    }
    __syncthreads();
  }

  // ---- normalize and store f32 attention output to workspace
  float inv[8];
#pragma unroll
  for (int r = 0; r < 8; ++r) inv[r] = __builtin_amdgcn_rcpf(l_i[r]);
#pragma unroll
  for (int j = 0; j < 4; ++j) {
#pragma unroll
    for (int r = 0; r < 8; ++r) {
      const int qrow = q0 + r + 8 * half;
      attn_out[((size_t)b * T_ + qrow) * C_ + hoff + j * 16 + ln] = o[j][r] * inv[r];
    }
  }
}

// ---------------------------------------------------------------------------
// Output projection: out = X @ W^T + bias.  M=B*T=8192, N=K=1024.
// One wave computes a 16x128 strip (8 accumulators), block covers 128x128.
// ---------------------------------------------------------------------------
__global__ __launch_bounds__(256)
void proj_kernel(const float* __restrict__ X, const float* __restrict__ W,
                 const float* __restrict__ bias, float* __restrict__ out)
{
  const int lane = threadIdx.x & 31;
  const int wv   = threadIdx.x >> 5;
  const int half = lane >> 4;
  const int ln   = lane & 15;
  const int m0   = blockIdx.y * 128 + wv * 16;
  const int n0b  = blockIdx.x * 128;

  v8f acc[8] = {};

#pragma unroll 1
  for (int kc = 0; kc < C_; kc += 32) {
    v16bf aX;   // A fragment from X (rows m0..m0+15, K chunk kc..kc+31)
#pragma unroll
    for (int v = 0; v < 8; ++v) {
      const int kk = 2 * v + 16 * half;
      const float* p = X + (size_t)(m0 + ln) * C_ + kc + kk;
      aX[2 * v]     = (__bf16)p[0];
      aX[2 * v + 1] = (__bf16)p[1];
    }
#pragma unroll
    for (int nc = 0; nc < 8; ++nc) {
      v16bf bW;   // B[k][n] = W[n][k] : contiguous in k per lane
#pragma unroll
      for (int v = 0; v < 8; ++v) {
        const int kk = 2 * v + 16 * half;
        const float* p = W + (size_t)(n0b + nc * 16 + ln) * C_ + kc + kk;
        bW[2 * v]     = (__bf16)p[0];
        bW[2 * v + 1] = (__bf16)p[1];
      }
      acc[nc] = __builtin_amdgcn_wmma_f32_16x16x32_bf16(false, aX, false, bW,
                                                        (short)0, acc[nc], false, false);
    }
  }

#pragma unroll
  for (int nc = 0; nc < 8; ++nc) {
    const float bv = bias[n0b + nc * 16 + ln];
#pragma unroll
    for (int r = 0; r < 8; ++r) {
      out[(size_t)(m0 + r + 8 * half) * C_ + n0b + nc * 16 + ln] = acc[nc][r] + bv;
    }
  }
}

extern "C" void kernel_launch(void* const* d_in, const int* in_sizes, int n_in,
                              void* d_out, int out_size, void* d_ws, size_t ws_size,
                              hipStream_t stream) {
  const float* q    = (const float*)d_in[0];
  const float* k    = (const float*)d_in[1];
  const float* v    = (const float*)d_in[2];
  const float* W    = (const float*)d_in[3];
  const float* bias = (const float*)d_in[4];
  float* out  = (float*)d_out;
  float* attn = (float*)d_ws;   // B*T*C f32 = 32 MiB scratch

  dim3 gA(T_ / 128, H_, B_);            // (16, 16, 4)
  flash_attn_kernel<<<gA, 256, 0, stream>>>(q, k, v, attn);

  dim3 gP(C_ / 128, (B_ * T_) / 128);   // (8, 64)
  proj_kernel<<<gP, 256, 0, stream>>>(attn, W, bias, out);
}
